// LinearRecurrenceLayer_49031346651685
// MI455X (gfx1250) — matrix operand order repaired and checked by reading, not compiled, not run on hardware
//
#include <hip/hip_runtime.h>
#include <hip/hip_bf16.h>
#include <math.h>

typedef __bf16 bf16_t;
typedef __attribute__((ext_vector_type(16))) __bf16 bf16x16;
typedef __attribute__((ext_vector_type(8)))  float  f32x8;
typedef int v4i __attribute__((vector_size(16)));

#define D_DIM 1024
#define L_DIM 4096
#define B_DIM 8
#define M_DIM (B_DIM * L_DIM)   /* 32768 rows */
#define NC 16                   /* scan chunks */
#define CL (L_DIM / NC)         /* 256 steps per chunk */

#define BM 128
#define BN 128
#define BK 32
#define LDA 40                  /* padded LDS row stride in bf16 elems (80B: kills 4-way bank conflicts) */

// gfx1250 async global->LDS path (ASYNCcnt-tracked, bypasses VGPRs)
#if __has_builtin(__builtin_amdgcn_global_load_async_to_lds_b128)
#define USE_ASYNC_LDS 1
#else
#define USE_ASYNC_LDS 0
#endif

#if __has_builtin(__builtin_amdgcn_s_wait_asynccnt)
#define WAIT_ASYNC0() __builtin_amdgcn_s_wait_asynccnt(0)
#else
#define WAIT_ASYNC0() asm volatile("s_wait_asynccnt 0" ::: "memory")
#endif

typedef __attribute__((address_space(1))) v4i* gv4i_p;   // global int4*
typedef __attribute__((address_space(3))) v4i* lv4i_p;   // LDS int4*

// ------------------------------------------------------------------
// f32 -> bf16 elementwise convert (weights)
// ------------------------------------------------------------------
__global__ __launch_bounds__(256) void cvt_f32_bf16(const float* __restrict__ src,
                                                    bf16_t* __restrict__ dst, int n) {
  int i = blockIdx.x * 256 + threadIdx.x;
  if (i < n) dst[i] = (bf16_t)src[i];
}

// ------------------------------------------------------------------
// RMSNorm row kernel: one 256-thread block per row, 4 f32 per thread.
// Produces bf16 xn for the WMMA GEMM.
// ------------------------------------------------------------------
__global__ __launch_bounds__(256) void rmsnorm_bf16(const float* __restrict__ x,
                                                    const float* __restrict__ scale,
                                                    bf16_t* __restrict__ xn) {
  const int row = blockIdx.x;
  const int tid = threadIdx.x;
  const float4 xv = *(const float4*)(x + (size_t)row * D_DIM + tid * 4);
  float ss = xv.x * xv.x + xv.y * xv.y + xv.z * xv.z + xv.w * xv.w;
  // wave32 reduce
  for (int off = 16; off > 0; off >>= 1) ss += __shfl_xor(ss, off, 32);
  __shared__ float red[8];
  __shared__ float s_inv;
  const int lane = tid & 31, wid = tid >> 5;
  if (lane == 0) red[wid] = ss;
  __syncthreads();
  if (tid == 0) {
    float tot = 0.f;
    for (int i = 0; i < 8; ++i) tot += red[i];
    float norm = sqrtf(tot / (float)D_DIM);   // ||x|| * D^-1/2
    s_inv = 1.0f / (norm + 1e-6f);
  }
  __syncthreads();
  const float inv = s_inv;
  const float4 sc = *(const float4*)(scale + tid * 4);
  union { bf16_t b[4]; uint2 u; } pk;
  pk.b[0] = (bf16_t)(sc.x * xv.x * inv);
  pk.b[1] = (bf16_t)(sc.y * xv.y * inv);
  pk.b[2] = (bf16_t)(sc.z * xv.z * inv);
  pk.b[3] = (bf16_t)(sc.w * xv.w * inv);
  *(uint2*)(xn + (size_t)row * D_DIM + tid * 4) = pk.u;
}

// ------------------------------------------------------------------
// bf16 WMMA GEMM: C[M,N] = A[M,K] * W[N,K]^T (+ epilogue)
// Block tile 128x128x32, 8 waves (4x2), each wave 32x64 = 8 WMMA accs.
// Double-buffered LDS; async global->LDS copies overlap with WMMA.
// mode 0: N=3072 gvf epilogue (sigmoid g -> bf16; v f32; sigmoid(f-1) f32)
// mode 1: N=1024 residual-add epilogue -> f32 d_out
// ------------------------------------------------------------------
__global__ __launch_bounds__(256) void wmma_gemm(
    const bf16_t* __restrict__ A, const bf16_t* __restrict__ W,
    int K, int tiles_n, int mode,
    bf16_t* __restrict__ g_buf, float* __restrict__ v_buf, float* __restrict__ f_buf,
    const float* __restrict__ resid, float* __restrict__ out) {
  __shared__ __align__(16) bf16_t sA[2][BM * LDA];
  __shared__ __align__(16) bf16_t sW[2][BN * LDA];

  const int tid    = threadIdx.x;
  const int lane   = tid & 31;
  const int wid    = tid >> 5;
  const int wave_m = wid >> 1;       // 0..3 -> 32-row slabs
  const int wave_n = wid & 1;        // 0..1 -> 64-col slabs
  const int row0   = (blockIdx.x / tiles_n) * BM;
  const int col0   = (blockIdx.x % tiles_n) * BN;
  const int m_lane = lane & 15;
  const int khalf  = lane >> 4;

  // issue one 128x32 bf16 tile of A and W into LDS buffer `buf`
  auto issue_tile = [&](int k0, int buf) {
#pragma unroll
    for (int i = 0; i < 2; ++i) {
      const int slot = tid + i * 256;   // 0..511
      const int r    = slot >> 2;       // 0..127
      const int c8   = (slot & 3) * 8;  // bf16 column
      const bf16_t* ga = A + (size_t)(row0 + r) * K + k0 + c8;
      const bf16_t* gw = W + (size_t)(col0 + r) * K + k0 + c8;
      bf16_t* la = &sA[buf][r * LDA + c8];
      bf16_t* lw = &sW[buf][r * LDA + c8];
#if USE_ASYNC_LDS
      __builtin_amdgcn_global_load_async_to_lds_b128((gv4i_p)(bf16_t*)ga, (lv4i_p)la, 0, 0);
      __builtin_amdgcn_global_load_async_to_lds_b128((gv4i_p)(bf16_t*)gw, (lv4i_p)lw, 0, 0);
#else
      *(uint4*)la = *(const uint4*)ga;
      *(uint4*)lw = *(const uint4*)gw;
#endif
    }
  };

  f32x8 acc[2][4];
  const f32x8 vzero = {0.f, 0.f, 0.f, 0.f, 0.f, 0.f, 0.f, 0.f};
#pragma unroll
  for (int s = 0; s < 2; ++s)
#pragma unroll
    for (int t = 0; t < 4; ++t) acc[s][t] = vzero;

  int cur = 0;
  issue_tile(0, 0);

  for (int k0 = 0; k0 < K; k0 += BK) {
#if USE_ASYNC_LDS
    WAIT_ASYNC0();
#endif
    __syncthreads();                       // tile `cur` visible to all waves
    if (k0 + BK < K) issue_tile(k0 + BK, cur ^ 1);  // stream next tile during compute

    // A fragment: lane (m, khalf) holds K = khalf*8+0..7 and 16+khalf*8+0..7  (ISA 7.12.2)
    bf16x16 afrag[2], wfrag[4];
#pragma unroll
    for (int s = 0; s < 2; ++s) {
      const int r = wave_m * 32 + s * 16 + m_lane;
      union { uint4 u[2]; bf16x16 v; } cv;
      cv.u[0] = *(const uint4*)(&sA[cur][r * LDA + khalf * 8]);
      cv.u[1] = *(const uint4*)(&sA[cur][r * LDA + khalf * 8 + 16]);
      afrag[s] = cv.v;
    }
    // B fragment: lane (n, khalf) holds K = khalf*8+0..7 and 16+khalf*8+0..7 of column n == W row n
#pragma unroll
    for (int t = 0; t < 4; ++t) {
      const int r = wave_n * 64 + t * 16 + m_lane;
      union { uint4 u[2]; bf16x16 v; } cv;
      cv.u[0] = *(const uint4*)(&sW[cur][r * LDA + khalf * 8]);
      cv.u[1] = *(const uint4*)(&sW[cur][r * LDA + khalf * 8 + 16]);
      wfrag[t] = cv.v;
    }

#pragma unroll
    for (int s = 0; s < 2; ++s)
#pragma unroll
      for (int t = 0; t < 4; ++t)
        acc[s][t] = __builtin_amdgcn_wmma_f32_16x16x32_bf16(
            false, afrag[s], false, wfrag[t], (short)0, acc[s][t], false, false);

    __syncthreads();                       // all waves done reading tile `cur`
    cur ^= 1;
  }

  // epilogue: D layout = lane(L) vgpr(r): row = r + 8*(L>>4), col = L&15
#pragma unroll
  for (int s = 0; s < 2; ++s) {
    const int rbase = row0 + wave_m * 32 + s * 16 + khalf * 8;
#pragma unroll
    for (int t = 0; t < 4; ++t) {
      const int e = col0 + wave_n * 64 + t * 16 + m_lane;
#pragma unroll
      for (int r = 0; r < 8; ++r) {
        const int m = rbase + r;
        const float val = acc[s][t][r];
        if (mode == 0) {
          if (e < D_DIM) {
            g_buf[(size_t)m * D_DIM + e] = (bf16_t)(1.0f / (1.0f + expf(-val)));
          } else if (e < 2 * D_DIM) {
            v_buf[(size_t)m * D_DIM + (e - D_DIM)] = val;
          } else {
            f_buf[(size_t)m * D_DIM + (e - 2 * D_DIM)] = 1.0f / (1.0f + expf(-(val - 1.0f)));
          }
        } else {
          const size_t idx = (size_t)m * D_DIM + e;
          out[idx] = resid[idx] + val;
        }
      }
    }
  }
}

// ------------------------------------------------------------------
// Chunked linear recurrence: h_t = f_t h_{t-1} + (1-f_t) v_t.
// Pass 1: per-chunk (P = prod f, S = local end-state from h=0)
// ------------------------------------------------------------------
__global__ __launch_bounds__(256) void scan_partial(const float* __restrict__ f_buf,
                                                    const float* __restrict__ v_buf,
                                                    float* __restrict__ P,
                                                    float* __restrict__ S) {
  const int dblk = blockIdx.x & 3;              // D/256 = 4
  const int c    = (blockIdx.x >> 2) & (NC - 1);
  const int b    = blockIdx.x >> 6;
  const int d    = dblk * 256 + threadIdx.x;
  float h = 0.f, p = 1.f;
  size_t base = ((size_t)b * L_DIM + (size_t)c * CL) * D_DIM + d;
  for (int t = 0; t < CL; ++t) {
    __builtin_prefetch(f_buf + base + 8 * D_DIM, 0, 3);
    __builtin_prefetch(v_buf + base + 8 * D_DIM, 0, 3);
    const float fv = f_buf[base];
    const float vv = v_buf[base];
    h = fv * h + (1.f - fv) * vv;
    p *= fv;
    base += D_DIM;
  }
  const size_t o = ((size_t)b * NC + c) * D_DIM + d;
  P[o] = p;
  S[o] = h;
}

// Pass 2: tiny sequential combine across chunks -> incoming state per chunk
__global__ __launch_bounds__(256) void scan_combine(const float* __restrict__ P,
                                                    const float* __restrict__ S,
                                                    float* __restrict__ H0) {
  const int dblk = blockIdx.x & 3;
  const int b    = blockIdx.x >> 2;
  const int d    = dblk * 256 + threadIdx.x;
  float h = 0.f;
  for (int c = 0; c < NC; ++c) {
    const size_t o = ((size_t)b * NC + c) * D_DIM + d;
    H0[o] = h;
    h = P[o] * h + S[o];
  }
}

// Pass 3: replay chunk with true incoming state, fuse y = g * h (bf16 out for GEMM2)
__global__ __launch_bounds__(256) void scan_apply(const float* __restrict__ f_buf,
                                                  const float* __restrict__ v_buf,
                                                  const bf16_t* __restrict__ g_buf,
                                                  const float* __restrict__ H0,
                                                  bf16_t* __restrict__ y_buf) {
  const int dblk = blockIdx.x & 3;
  const int c    = (blockIdx.x >> 2) & (NC - 1);
  const int b    = blockIdx.x >> 6;
  const int d    = dblk * 256 + threadIdx.x;
  float h = H0[((size_t)b * NC + c) * D_DIM + d];
  size_t base = ((size_t)b * L_DIM + (size_t)c * CL) * D_DIM + d;
  for (int t = 0; t < CL; ++t) {
    __builtin_prefetch(f_buf + base + 8 * D_DIM, 0, 3);
    __builtin_prefetch(v_buf + base + 8 * D_DIM, 0, 3);
    const float fv = f_buf[base];
    const float vv = v_buf[base];
    h = fv * h + (1.f - fv) * vv;
    const float g = (float)g_buf[base];
    y_buf[base] = (bf16_t)(g * h);
    base += D_DIM;
  }
}

// ------------------------------------------------------------------
extern "C" void kernel_launch(void* const* d_in, const int* in_sizes, int n_in,
                              void* d_out, int out_size, void* d_ws, size_t ws_size,
                              hipStream_t stream) {
  const float* x     = (const float*)d_in[0];
  const float* w_in  = (const float*)d_in[1];
  const float* w_out = (const float*)d_in[2];
  const float* scale = (const float*)d_in[3];
  float* out = (float*)d_out;

  char* ws = (char*)d_ws;
  size_t off = 0;
  auto alloc = [&](size_t bytes) -> void* {
    void* p = ws + off;
    off = (off + bytes + 255) & ~(size_t)255;
    return p;
  };

  bf16_t* xn   = (bf16_t*)alloc((size_t)M_DIM * D_DIM * 2);
  bf16_t* wib  = (bf16_t*)alloc((size_t)3 * D_DIM * D_DIM * 2);
  bf16_t* wob  = (bf16_t*)alloc((size_t)D_DIM * D_DIM * 2);
  bf16_t* g    = (bf16_t*)alloc((size_t)M_DIM * D_DIM * 2);
  float*  v    = (float*) alloc((size_t)M_DIM * D_DIM * 4);
  float*  f    = (float*) alloc((size_t)M_DIM * D_DIM * 4);
  bf16_t* y    = (bf16_t*)alloc((size_t)M_DIM * D_DIM * 2);
  float*  P    = (float*) alloc((size_t)B_DIM * NC * D_DIM * 4);
  float*  S    = (float*) alloc((size_t)B_DIM * NC * D_DIM * 4);
  float*  H0   = (float*) alloc((size_t)B_DIM * NC * D_DIM * 4);
  (void)ws_size; (void)in_sizes; (void)n_in; (void)out_size;

  // weight conversion to bf16
  {
    int n1 = 3 * D_DIM * D_DIM;
    cvt_f32_bf16<<<(n1 + 255) / 256, 256, 0, stream>>>(w_in, wib, n1);
    int n2 = D_DIM * D_DIM;
    cvt_f32_bf16<<<(n2 + 255) / 256, 256, 0, stream>>>(w_out, wob, n2);
  }

  // RMSNorm + bf16 cast: one block per row
  rmsnorm_bf16<<<M_DIM, 256, 0, stream>>>(x, scale, xn);

  // GEMM1: [32768,1024] x [3072,1024]^T -> fused sigmoid gates
  wmma_gemm<<<(M_DIM / BM) * (3 * D_DIM / BN), 256, 0, stream>>>(
      xn, wib, D_DIM, 3 * D_DIM / BN, 0, g, v, f, nullptr, nullptr);

  // chunked linear recurrence
  scan_partial<<<B_DIM * NC * (D_DIM / 256), 256, 0, stream>>>(f, v, P, S);
  scan_combine<<<B_DIM * (D_DIM / 256), 256, 0, stream>>>(P, S, H0);
  scan_apply<<<B_DIM * NC * (D_DIM / 256), 256, 0, stream>>>(f, v, g, H0, y);

  // GEMM2: [32768,1024] x [1024,1024]^T + residual -> f32 out
  wmma_gemm<<<(M_DIM / BM) * (D_DIM / BN), 256, 0, stream>>>(
      y, wob, D_DIM, D_DIM / BN, 1, nullptr, nullptr, nullptr, x, out);
}
